// AttentionHead_455266533415
// MI455X (gfx1250) — matrix operand order repaired and checked by reading, not compiled, not run on hardware
//
#include <hip/hip_runtime.h>
#include <hip/hip_bf16.h>
#include <stdint.h>

#define E_DIM 2048
#define T_DIM 2048
#define H_DIM 128
#define NBATCH 8

typedef __attribute__((ext_vector_type(16))) __bf16 v16bf;
typedef __attribute__((ext_vector_type(8)))  float  v8f;

union FragB {
    v16bf v;
    uint4 q[2];
    __bf16 b[16];
};

__device__ __forceinline__ v8f wmma_bf16(v16bf a, v16bf b, v8f c) {
    // D = A(16x32) * B(32x16) + C(16x16 f32)
    return __builtin_amdgcn_wmma_f32_16x16x32_bf16(
        /*neg_a=*/false, a, /*neg_b=*/false, b,
        /*c_mod=*/(short)0, c, /*reuse_a=*/false, /*reuse_b=*/false);
}

#define SM_SCALE 0.08838834764831845f   // 1/sqrt(128)

// ---------------------------------------------------------------------------
// Kernel 1: transpose + convert weights to bf16.  Wq pre-scaled by 1/sqrt(H).
// W : [E,H] f32 row-major  ->  WT : [H,E] bf16 row-major (Q,K,V concatenated)
// ---------------------------------------------------------------------------
__global__ __launch_bounds__(256)
void attn_prep_kernel(const float* __restrict__ Wq, const float* __restrict__ Wk,
                      const float* __restrict__ Wv, __bf16* __restrict__ WT) {
    int i = blockIdx.x * 256 + threadIdx.x;      // over E*H
    if (i >= E_DIM * H_DIM) return;
    int e = i / H_DIM;
    int h = i % H_DIM;
    size_t o = (size_t)h * E_DIM + e;
    WT[o]                             = (__bf16)(Wq[i] * SM_SCALE);
    WT[(size_t)E_DIM * H_DIM + o]     = (__bf16)Wk[i];
    WT[(size_t)2 * E_DIM * H_DIM + o] = (__bf16)Wv[i];
}

// ---------------------------------------------------------------------------
// Kernel 2: QKV projection GEMM via WMMA bf16.
// wid%3 = matrix (Q/K/V waves for same token tile run adjacently -> x read
// from HBM once).  K-loop unrolled by 2 with alternating register sets; B
// fragments flow through a 3-deep software pipeline so every WMMA has >=6
// b128 loads in flight ahead of it.
// ---------------------------------------------------------------------------
__global__ __launch_bounds__(256)
void attn_qkv_kernel(const float* __restrict__ x, const __bf16* __restrict__ WT,
                     const float* __restrict__ bq, const float* __restrict__ bk,
                     const float* __restrict__ bv,
                     __bf16* __restrict__ Qb, __bf16* __restrict__ Kb,
                     __bf16* __restrict__ Vt) {
    const int lane = threadIdx.x & 31;
    const int wave = threadIdx.x >> 5;
    const int wid  = blockIdx.x * 8 + wave;      // 0..3071
    const int rt   = wid / 3;                    // 16-row tile over B*T
    const int mat  = wid - rt * 3;               // 0=Q 1=K 2=V
    const int m0   = rt * 16;
    const int hf   = lane >> 4;                  // lane half
    const int ln   = lane & 15;

    const __bf16* wt = WT + (size_t)mat * E_DIM * H_DIM;   // [H][E]
    const float* bias = (mat == 0) ? bq : ((mat == 1) ? bk : bv);
    const float bscale = (mat == 0) ? SM_SCALE : 1.0f;
    const float* xrow = x + (size_t)(m0 + ln) * E_DIM;

    v8f acc[8];
#pragma unroll
    for (int nt = 0; nt < 8; ++nt) acc[nt] = (v8f){0.f,0.f,0.f,0.f,0.f,0.f,0.f,0.f};

#define LOADB_(arr, i, kbase)                                                   \
    do {                                                                        \
        const __bf16* wp_ = wt + (size_t)((i) * 16 + ln) * E_DIM + (kbase) + hf * 8; \
        (arr)[i].q[0] = *(const uint4*)wp_;                                     \
        (arr)[i].q[1] = *(const uint4*)(wp_ + 16);                              \
    } while (0)

#define CVT_A_(a, g0, g1, g2, g3)                                               \
    do {                                                                        \
        (a).b[0] = (__bf16)(g0).x;  (a).b[1] = (__bf16)(g0).y;                  \
        (a).b[2] = (__bf16)(g0).z;  (a).b[3] = (__bf16)(g0).w;                  \
        (a).b[4] = (__bf16)(g1).x;  (a).b[5] = (__bf16)(g1).y;                  \
        (a).b[6] = (__bf16)(g1).z;  (a).b[7] = (__bf16)(g1).w;                  \
        (a).b[8] = (__bf16)(g2).x;  (a).b[9] = (__bf16)(g2).y;                  \
        (a).b[10] = (__bf16)(g2).z; (a).b[11] = (__bf16)(g2).w;                 \
        (a).b[12] = (__bf16)(g3).x; (a).b[13] = (__bf16)(g3).y;                 \
        (a).b[14] = (__bf16)(g3).z; (a).b[15] = (__bf16)(g3).w;                 \
    } while (0)

    FragB btA[8], btB[8];

    // Prologue: first A-tile raw loads + first 3 B fragments.
    float4 fa0 = *(const float4*)(xrow + hf * 8);
    float4 fa1 = *(const float4*)(xrow + hf * 8 + 4);
    float4 fa2 = *(const float4*)(xrow + hf * 8 + 16);
    float4 fa3 = *(const float4*)(xrow + hf * 8 + 20);
    LOADB_(btA, 0, 0); LOADB_(btA, 1, 0); LOADB_(btA, 2, 0);

    for (int kb = 0; kb < E_DIM; kb += 64) {
        // --- half 0: K-chunk kb ---
        {   // issue raw A loads for kb+32 (always in range)
            const float* nx = xrow + kb + 32 + hf * 8;
            float4 fb0 = *(const float4*)(nx);
            float4 fb1 = *(const float4*)(nx + 4);
            float4 fb2 = *(const float4*)(nx + 16);
            float4 fb3 = *(const float4*)(nx + 20);

            FragB a;
            CVT_A_(a, fa0, fa1, fa2, fa3);
#pragma unroll
            for (int nt = 0; nt < 8; ++nt) {
                if (nt < 5) LOADB_(btA, nt + 3, kb);
                else        LOADB_(btB, nt - 5, kb + 32);
                acc[nt] = wmma_bf16(a.v, btA[nt].v, acc[nt]);
            }

            // --- half 1: K-chunk kb+32 ---
            const int kn = (kb + 64 < E_DIM) ? (kb + 64) : 0;  // clamp: harmless
            const float* nx2 = xrow + kn + hf * 8;
            fa0 = *(const float4*)(nx2);
            fa1 = *(const float4*)(nx2 + 4);
            fa2 = *(const float4*)(nx2 + 16);
            fa3 = *(const float4*)(nx2 + 20);

            FragB a2;
            CVT_A_(a2, fb0, fb1, fb2, fb3);
#pragma unroll
            for (int nt = 0; nt < 8; ++nt) {
                if (nt < 5) LOADB_(btB, nt + 3, kb + 32);
                else        LOADB_(btA, nt - 5, kn);
                acc[nt] = wmma_bf16(a2.v, btB[nt].v, acc[nt]);
            }
        }
    }

    // bias add + store
#pragma unroll
    for (int nt = 0; nt < 8; ++nt) {
        const float bval = bias[nt * 16 + ln] * bscale;
#pragma unroll
        for (int v = 0; v < 8; ++v) {
            float r = acc[nt][v] + bval;
            int m = m0 + v + hf * 8;                  // global token row
            if (mat == 2) {
                int b = m >> 11;                      // / T_DIM
                int t = m & (T_DIM - 1);
                Vt[((size_t)b * H_DIM + nt * 16 + ln) * T_DIM + t] = (__bf16)r;
            } else {
                __bf16* dst = (mat == 0) ? Qb : Kb;
                dst[(size_t)m * H_DIM + nt * 16 + ln] = (__bf16)r;
            }
        }
    }
#undef LOADB_
#undef CVT_A_
}

// ---------------------------------------------------------------------------
// Kernel 3: causal flash attention.  One wave per (batch, 16-query tile).
// K fragments run through a 3-deep pipeline into 8 S-WMMAs; V fragment loads
// issue immediately after so they overlap the softmax VALU work and the LDS
// D-layout -> A-layout transpose of P; then 8 WMMAs for O += P*V.
// ---------------------------------------------------------------------------
__global__ __launch_bounds__(256)
void attn_fa_kernel(const __bf16* __restrict__ Qb,
                    const __bf16* __restrict__ Kb,
                    const __bf16* __restrict__ Vt,
                    float* __restrict__ out) {
    __shared__ __bf16 ldsP[8][16 * 32];   // 1KB per wave

    const int lane = threadIdx.x & 31;
    const int wave = threadIdx.x >> 5;
    const int wid  = blockIdx.x * 8 + wave;       // 0..1023
    const int b    = wid >> 7;                    // 128 q-tiles per batch
    const int qt   = wid & 127;
    const int q0   = qt * 16;
    const int hf   = lane >> 4;
    const int ln   = lane & 15;

    const __bf16* Qrow  = Qb + ((size_t)b * T_DIM + q0) * H_DIM;
    const __bf16* Kbase = Kb + (size_t)b * T_DIM * H_DIM;
    const __bf16* Vbase = Vt + (size_t)b * H_DIM * T_DIM;

    // Preload Q A-fragments for all 4 K-chunks (head dim = 128).
    FragB qa[4];
#pragma unroll
    for (int kf = 0; kf < 4; ++kf) {
        const int base = kf * 32 + hf * 8;
        qa[kf].q[0] = *(const uint4*)(Qrow + (size_t)ln * H_DIM + base);
        qa[kf].q[1] = *(const uint4*)(Qrow + (size_t)ln * H_DIM + base + 16);
    }

    float m_run[8], l_run[8];
    v8f o[8];
#pragma unroll
    for (int v = 0; v < 8; ++v) { m_run[v] = -3.0e38f; l_run[v] = 0.f; }
#pragma unroll
    for (int t = 0; t < 8; ++t) o[t] = (v8f){0.f,0.f,0.f,0.f,0.f,0.f,0.f,0.f};

#define LOADK_(f)                                                               \
    do {                                                                        \
        const int kf_ = (f) & 3;                                                \
        const int tl_ = (f) >> 2;                                               \
        const __bf16* kp_ =                                                     \
            Kbase + (size_t)(j0 + tl_ * 16 + ln) * H_DIM + kf_ * 32 + hf * 8;   \
        kfr[f].q[0] = *(const uint4*)kp_;                                       \
        kfr[f].q[1] = *(const uint4*)(kp_ + 16);                                \
    } while (0)

    const int nIter = (q0 + 47) >> 5;             // 32-key tiles up to diagonal
    for (int it = 0; it < nIter; ++it) {
        const int j0 = it * 32;

        // ---- S = Q * K^T : 3-deep pipelined K fragments, 8 WMMAs ----
        FragB kfr[8];                             // f<4: keys j0..+15, f>=4: +16..+31
        LOADK_(0); LOADK_(1); LOADK_(2);
        v8f s0 = (v8f){0.f,0.f,0.f,0.f,0.f,0.f,0.f,0.f};
        v8f s1 = s0;
#pragma unroll
        for (int f = 0; f < 8; ++f) {
            if (f < 5) LOADK_(f + 3);
            if (f < 4) s0 = wmma_bf16(qa[f].v,     kfr[f].v, s0);
            else       s1 = wmma_bf16(qa[f - 4].v, kfr[f].v, s1);
        }

        // ---- issue V-fragment loads now: overlap softmax + LDS transpose ----
        FragB vfr[8];
#pragma unroll
        for (int t = 0; t < 8; ++t) {
            const __bf16* vp =
                Vbase + (size_t)(t * 16 + ln) * T_DIM + j0 + hf * 8;
            vfr[t].q[0] = *(const uint4*)vp;
            vfr[t].q[1] = *(const uint4*)(vp + 16);
        }

        // ---- causal mask (scale already folded into Q) ----
#pragma unroll
        for (int v = 0; v < 8; ++v) {
            const int row  = q0 + v + hf * 8;
            const int key0 = j0 + ln;
            float a = s0[v];
            float c = s1[v];
            if (key0 > row)      a = -3.0e38f;
            if (key0 + 16 > row) c = -3.0e38f;
            s0[v] = a; s1[v] = c;
        }

        // ---- online softmax: row max / rescale / exp / row sum ----
        float alpha[8];
#pragma unroll
        for (int v = 0; v < 8; ++v) {
            float r = fmaxf(s0[v], s1[v]);
            r = fmaxf(r, __shfl_xor(r, 1, 32));
            r = fmaxf(r, __shfl_xor(r, 2, 32));
            r = fmaxf(r, __shfl_xor(r, 4, 32));
            r = fmaxf(r, __shfl_xor(r, 8, 32));
            const float mn = fmaxf(m_run[v], r);
            const float al = __expf(m_run[v] - mn);
            m_run[v] = mn;
            l_run[v] *= al;
            alpha[v] = al;
        }
#pragma unroll
        for (int t = 0; t < 8; ++t)
#pragma unroll
            for (int v = 0; v < 8; ++v) o[t][v] *= alpha[v];
#pragma unroll
        for (int v = 0; v < 8; ++v) {
            s0[v] = __expf(s0[v] - m_run[v]);
            s1[v] = __expf(s1[v] - m_run[v]);
            float rs = s0[v] + s1[v];
            rs += __shfl_xor(rs, 1, 32);
            rs += __shfl_xor(rs, 2, 32);
            rs += __shfl_xor(rs, 4, 32);
            rs += __shfl_xor(rs, 8, 32);
            l_run[v] += rs;
        }

        // ---- P (D-layout) -> LDS row-major [16][32] -> A-fragment ----
        __bf16* lp = ldsP[wave];
#pragma unroll
        for (int v = 0; v < 8; ++v) {
            const int row = v + hf * 8;
            lp[row * 32 + ln]      = (__bf16)s0[v];
            lp[row * 32 + 16 + ln] = (__bf16)s1[v];
        }
        FragB pa;
        {
            const int base = hf * 8;
            pa.q[0] = *(const uint4*)(lp + ln * 32 + base);
            pa.q[1] = *(const uint4*)(lp + ln * 32 + base + 16);
        }

        // ---- O += P * V ----
#pragma unroll
        for (int t = 0; t < 8; ++t)
            o[t] = wmma_bf16(pa.v, vfr[t].v, o[t]);
    }
#undef LOADK_

    // ---- epilogue: divide by running sum, store f32 ----
    float inv[8];
#pragma unroll
    for (int v = 0; v < 8; ++v) inv[v] = 1.0f / l_run[v];
#pragma unroll
    for (int t = 0; t < 8; ++t)
#pragma unroll
        for (int v = 0; v < 8; ++v) {
            out[((size_t)b * T_DIM + q0 + v + hf * 8) * H_DIM + t * 16 + ln] =
                o[t][v] * inv[v];
        }
}

// ---------------------------------------------------------------------------
extern "C" void kernel_launch(void* const* d_in, const int* in_sizes, int n_in,
                              void* d_out, int out_size, void* d_ws, size_t ws_size,
                              hipStream_t stream) {
    const float* x    = (const float*)d_in[0];
    const float* Wq_p = (const float*)d_in[1];
    const float* bq_p = (const float*)d_in[2];
    const float* Wk_p = (const float*)d_in[3];
    const float* bk_p = (const float*)d_in[4];
    const float* Wv_p = (const float*)d_in[5];
    const float* bv_p = (const float*)d_in[6];
    float* out = (float*)d_out;

    char* ws = (char*)d_ws;
    const size_t WT_ELEMS  = (size_t)3 * E_DIM * H_DIM;          // bf16
    const size_t QKV_ELEMS = (size_t)NBATCH * T_DIM * H_DIM;     // bf16 each
    __bf16* WT = (__bf16*)ws;
    __bf16* Qb = (__bf16*)(ws + WT_ELEMS * 2);
    __bf16* Kb = Qb + QKV_ELEMS;
    __bf16* Vt = Kb + QKV_ELEMS;
    // total ws use: 2*(3*E*H + 3*B*T*H) bytes ~= 13.5 MB

    attn_prep_kernel<<<(E_DIM * H_DIM + 255) / 256, 256, 0, stream>>>(
        Wq_p, Wk_p, Wv_p, WT);

    // 3 matrices * 1024 row tiles = 3072 waves, 8 waves/block
    attn_qkv_kernel<<<384, 256, 0, stream>>>(
        x, WT, bq_p, bk_p, bv_p, Qb, Kb, Vt);

    // 8 batches * 128 q-tiles = 1024 waves, 8 waves/block
    attn_fa_kernel<<<128, 256, 0, stream>>>(Qb, Kb, Vt, out);
}
